// AttentionBlock_15006615733924
// MI455X (gfx1250) — matrix operand order repaired and checked by reading, not compiled, not run on hardware
//
#include <hip/hip_runtime.h>
#include <cstdint>
#include <cstddef>

// CDNA5 / gfx1250, wave32. WMMA fragment types:
typedef _Float16 half16 __attribute__((ext_vector_type(16))); // 8 VGPRs of packed f16
typedef float    float8 __attribute__((ext_vector_type(8)));  // 16x16 f32 C/D tile

#define NB 32    // batch
#define NC 32    // channels (== heads*d, H=1, d=32)
#define LL 511   // sequence length
#define LP 512   // padded length
#define RW 1024  // padded row width for shift-image (255 zeros + 511 data + pad)

// ---- gfx1250 async global->LDS copy (ASYNCcnt path), guarded fallback ----
#if defined(__HIP_DEVICE_COMPILE__) && __has_builtin(__builtin_amdgcn_global_load_async_to_lds_b32)
#define USE_ASYNC_LDS 1
typedef __attribute__((address_space(1))) int as1_int;
typedef __attribute__((address_space(3))) int as3_int;
__device__ __forceinline__ void async_copy_b32(const void* gsrc, void* ldst) {
  __builtin_amdgcn_global_load_async_to_lds_b32(
      (as1_int*)(unsigned long long)gsrc,
      (as3_int*)(unsigned int)(unsigned long long)ldst, 0, 0);
}
__device__ __forceinline__ void async_wait0() {
#if __has_builtin(__builtin_amdgcn_s_wait_asynccnt)
  __builtin_amdgcn_s_wait_asynccnt(0);
#else
  asm volatile("s_wait_asynccnt 0" ::: "memory");
#endif
}
#endif

// ---------------- kernel: zero the padded f16 image ----------------
__global__ void zero_f16_kernel(float4* __restrict__ p, int n4) {
  int i = blockIdx.x * blockDim.x + threadIdx.x;
  if (i < n4) p[i] = make_float4(0.f, 0.f, 0.f, 0.f);
}

// ---------------- kernel: group norm (G=32 -> one channel per group) ----
__global__ void gn_kernel(const float* __restrict__ x, const float* __restrict__ w,
                          const float* __restrict__ bb, float* __restrict__ h) {
  int row = blockIdx.x;          // b*32 + ch, 1024 rows
  int ch  = row & 31;
  const float* xr = x + (size_t)row * LL;
  float s = 0.f, s2 = 0.f;
  for (int i = threadIdx.x; i < LL; i += 256) { float v = xr[i]; s += v; s2 += v * v; }
  __shared__ float r0[256], r1[256];
  r0[threadIdx.x] = s; r1[threadIdx.x] = s2; __syncthreads();
  for (int off = 128; off > 0; off >>= 1) {
    if (threadIdx.x < off) { r0[threadIdx.x] += r0[threadIdx.x + off];
                             r1[threadIdx.x] += r1[threadIdx.x + off]; }
    __syncthreads();
  }
  float mu  = r0[0] * (1.0f / LL);
  float var = r1[0] * (1.0f / LL) - mu * mu;
  float inv = rsqrtf(var + 1e-5f);
  float sc = w[ch] * inv;
  float sb = bb[ch] - mu * sc;
  float* hr = h + (size_t)row * LL;
  for (int i = threadIdx.x; i < LL; i += 256) hr[i] = xr[i] * sc + sb;
}

// ---------------- kernel: qkv 1x1 conv -> f16 q (pre-scaled), k, v ------
__global__ void qkv_kernel(const float* __restrict__ h, const float* __restrict__ w,
                           const float* __restrict__ bias,
                           _Float16* __restrict__ qh, _Float16* __restrict__ kh,
                           _Float16* __restrict__ vh) {
  int idx = blockIdx.x * blockDim.x + threadIdx.x;      // 32*96*512
  if (idx >= NB * 96 * LP) return;
  int l = idx & (LP - 1);
  int o = (idx >> 9) % 96;
  int b = idx / (96 * LP);
  int ch = o & 31, which = o >> 5;
  _Float16* dst = (which == 0) ? qh : (which == 1) ? kh : vh;
  size_t di = ((size_t)b * NC + ch) * LP + l;
  if (l == LL) { dst[di] = (_Float16)0.f; return; }     // pad column -> 0
  float acc = bias[o];
  const float* hb = h + (size_t)b * NC * LL + l;
  const float* wr = w + o * NC;
  #pragma unroll
  for (int c = 0; c < NC; c++) acc += wr[c] * hb[(size_t)c * LL];
  if (which == 0) acc *= 0.17677669529663687f;          // d^-0.5 folded into q
  dst[di] = (_Float16)acc;
}

// ---------------- kernel: S = q^T k  (WMMA f16, K = d = 32) -------------
__global__ void s_kernel(const _Float16* __restrict__ qh, const _Float16* __restrict__ kh,
                         float* __restrict__ S) {
  int wid  = blockIdx.x * 4 + (threadIdx.x >> 5);       // 32768 tiles
  int lane = threadIdx.x & 31;
  int b    = wid >> 10;
  int tile = wid & 1023;
  int nt = tile >> 5, mt = tile & 31;
  const _Float16* q = qh + (size_t)b * NC * LP;
  const _Float16* k = kh + (size_t)b * NC * LP;
  int hl = lane & 15, hi = lane >> 4;
  int n = nt * 16 + hl;
  int m = mt * 16 + hl;
  half16 a, bf;
  #pragma unroll
  for (int v = 0; v < 8; v++) {
    int da = (hi ? 16 : 0) + 2 * v;                            // A: lane=row n, K=d
    a[2 * v]     = q[(size_t)da * LP + n];
    a[2 * v + 1] = q[(size_t)(da + 1) * LP + n];
    int db = (hi ? 16 : 0) + 2 * v;                            // B: lane=col m, K=d
    bf[2 * v]     = k[(size_t)db * LP + m];
    bf[2 * v + 1] = k[(size_t)(db + 1) * LP + m];
  }
  float8 c = {};
  c = __builtin_amdgcn_wmma_f32_16x16x32_f16(false, a, false, bf, (short)0, c, false, false);
  float* Sb = S + (size_t)b * LP * LP;
  #pragma unroll
  for (int r = 0; r < 8; r++) {
    int row = nt * 16 + r + (hi ? 8 : 0);
    Sb[(size_t)row * LP + mt * 16 + hl] = c[r];
  }
}

// ---------------- kernel: row softmax -> f16 attn (padded col = 0) ------
__global__ void softmax_kernel(const float* __restrict__ S, _Float16* __restrict__ attnh) {
  int row = blockIdx.x;                                  // b*512 + n
  int n = row & (LP - 1);
  _Float16* dst = attnh + (size_t)row * LP;
  if (n == LL) {                                         // pad row -> zeros
    for (int i = threadIdx.x; i < LP; i += 128) dst[i] = (_Float16)0.f;
    return;
  }
  const float* src = S + (size_t)row * LP;
  __shared__ float red[128];
  float mx = -1e30f;
  for (int i = threadIdx.x; i < LL; i += 128) mx = fmaxf(mx, src[i]);
  red[threadIdx.x] = mx; __syncthreads();
  for (int off = 64; off > 0; off >>= 1) {
    if (threadIdx.x < off) red[threadIdx.x] = fmaxf(red[threadIdx.x], red[threadIdx.x + off]);
    __syncthreads();
  }
  mx = red[0]; __syncthreads();
  float sum = 0.f;
  for (int i = threadIdx.x; i < LL; i += 128) sum += __expf(src[i] - mx);
  red[threadIdx.x] = sum; __syncthreads();
  for (int off = 64; off > 0; off >>= 1) {
    if (threadIdx.x < off) red[threadIdx.x] += red[threadIdx.x + off];
    __syncthreads();
  }
  float inv = 1.0f / red[0];
  for (int i = threadIdx.x; i < LP; i += 128) {
    float v = (i < LL) ? __expf(src[i] - mx) * inv : 0.0f;
    dst[i] = (_Float16)v;
  }
}

// ------- kernel: out = attn * v^T (WMMA, K=512) -> padded shift image ----
__global__ void av_kernel(const _Float16* __restrict__ attnh, const _Float16* __restrict__ vh,
                          _Float16* __restrict__ Ash) {
  int wid  = blockIdx.x * 4 + (threadIdx.x >> 5);        // 2048 tiles
  int lane = threadIdx.x & 31;
  int b   = wid >> 6;
  int rem = wid & 63;
  int nt = rem >> 1, dt = rem & 1;
  const _Float16* Ab = attnh + (size_t)b * LP * LP;
  const _Float16* Vb = vh + (size_t)b * NC * LP;
  int hl = lane & 15, hi = lane >> 4;
  int n = nt * 16 + hl;                                  // A row
  int d = dt * 16 + hl;                                  // B col
  float8 c = {};
  for (int kt = 0; kt < 16; kt++) {
    half16 a, bf;
    const uint32_t* arow = (const uint32_t*)(Ab + (size_t)n * LP + kt * 32);
    #pragma unroll
    for (int v = 0; v < 8; v++) {
      int ma = (hi ? 16 : 0) + 2 * v;                                    // A m-pair
      ((uint32_t*)&a)[v] = arow[ma >> 1];
      int mb = kt * 32 + (hi ? 16 : 0) + 2 * v;                          // B m-pair
      ((uint32_t*)&bf)[v] = *(const uint32_t*)(Vb + (size_t)d * LP + mb);
    }
    c = __builtin_amdgcn_wmma_f32_16x16x32_f16(false, a, false, bf, (short)0, c, false, false);
  }
  // C[M=n][N=d] -> Ash[b][d][256 + n]   (f16 shift image)
  _Float16* AshB = Ash + (size_t)b * NC * RW;
  #pragma unroll
  for (int r = 0; r < 8; r++) {
    int nn = nt * 16 + r + (hi ? 8 : 0);
    if (nn < LL) AshB[(size_t)d * RW + 256 + nn] = (_Float16)c[r];
  }
}

// ------- kernel: pre-swizzle the unshifted operand into WMMA-A layout ----
__global__ void aswz_kernel(const _Float16* __restrict__ Ash, uint32_t* __restrict__ Aswz) {
  int t = blockIdx.x * blockDim.x + threadIdx.x;         // 2*512*32
  if (t >= 2 * 512 * 32) return;
  int lane = t & 31;
  int kt   = (t >> 5) & 511;
  int ot   = t >> 14;
  int hl = lane & 15, hi = lane >> 4;
  int o = ot * 16 + hl;                                  // A-matrix row = weight batch idx
  uint32_t* dst = Aswz + (size_t)t * 8;
  #pragma unroll
  for (int v = 0; v < 8; v++) {
    int jb = kt * 32 + (hi ? 16 : 0) + 2 * v;
    union { _Float16 h[2]; uint32_t u; } pk;
    #pragma unroll
    for (int e = 0; e < 2; e++) {
      int j = jb + e;
      int i = j >> 9;                                    // channel
      int k = j & 511;                                   // position (511 == pad, slot holds 0)
      pk.h[e] = Ash[((size_t)o * NC + i) * RW + 256 + k];
    }
    dst[v] = pk.u;
  }
}

// ------- kernel: self-conv as 511 shift-GEMMs (32x32, K=16384, WMMA) -----
// Per block: one shift s. Per channel i: async-stage the shifted 32-row
// window into LDS (4-byte aligned via even shift; odd correction done with
// register funnel shifts), then 16 WMMA k-steps fed by ds_load_b128.
__global__ void __launch_bounds__(128)
selfconv_kernel(const _Float16* __restrict__ Ash, const uint32_t* __restrict__ Aswz,
                float* __restrict__ y) {
  __shared__ uint32_t ldsW[32 * 260];           // 32 n-rows x 257 (+3 pad) u32

  int t    = blockIdx.x;                        // output position
  int s    = t - 255;                           // shift
  int sodd = s & 1;
  int s2   = s - sodd;                          // even part of shift (halves)
  int w    = threadIdx.x >> 5;                  // 4 waves: 2x2 output tiles
  int lane = threadIdx.x & 31;
  int ot = w & 1, nt = w >> 1;
  int hl = lane & 15, hi = lane >> 4;
  int n = nt * 16 + hl;                         // B column = input batch idx

  // staging assignment: 4 threads per n-row
  int sr = threadIdx.x >> 2;                    // 0..31 (n-row)
  int sc = threadIdx.x & 3;
  const uint32_t* AshU = (const uint32_t*)Ash;  // 512 u32 per (n,i) padded row
  uint32_t* ldsRow = ldsW + sr * 260;
  int gshift = 128 + (s2 >> 1);                 // u32 offset of window start (>= 0)

  const uint32_t* crow = ldsW + (size_t)n * 260;          // consumer row
  const uint4* abase = (const uint4*)Aswz + (size_t)ot * 512 * 64 + lane * 2;

  float8 c = {};
  for (int i = 0; i < NC; ++i) {
    __syncthreads();                            // WAR: prior consumers done
    const uint32_t* gsrc = AshU + ((size_t)(sr * NC + i) << 9) + gshift;
#ifdef USE_ASYNC_LDS
    for (int m = sc; m < 257; m += 4) async_copy_b32(gsrc + m, ldsRow + m);
    async_wait0();
#else
    for (int m = sc; m < 257; m += 4) ldsRow[m] = gsrc[m];
#endif
    __syncthreads();
    #pragma unroll 4
    for (int kt = 0; kt < 16; ++kt) {
      union { uint4 u[2]; half16 h; } A;        // pre-swizzled A fragment, 2x b128
      A.u[0] = abase[(size_t)(i * 16 + kt) * 64];
      A.u[1] = abase[(size_t)(i * 16 + kt) * 64 + 1];
      int u0 = kt * 16 + hi * 8;                // u32 index of this lane's B chunk
      uint32_t U[9];
      *(uint4*)(U)     = *(const uint4*)(crow + u0);      // ds_load_b128
      *(uint4*)(U + 4) = *(const uint4*)(crow + u0 + 4);  // ds_load_b128
      U[8] = crow[u0 + 8];
      half16 bf;
      uint32_t* bw = (uint32_t*)&bf;
      #pragma unroll
      for (int v = 0; v < 8; ++v)               // odd-shift correction (uniform branch)
        bw[v] = sodd ? ((U[v] >> 16) | (U[v + 1] << 16)) : U[v];
      // NB: k==511 pad needs no masking here: the A operand is exactly 0 there.
      c = __builtin_amdgcn_wmma_f32_16x16x32_f16(false, A.h, false, bf, (short)0, c, false, false);
    }
  }
  #pragma unroll
  for (int r = 0; r < 8; ++r) {
    int o = ot * 16 + r + (hi ? 8 : 0);         // C[M=o][N=n] -> y[n][o][t]
    y[((size_t)n * NC + o) * LL + t] = c[r];
  }
}

// ---------------- kernel: proj 1x1 conv + bias + residual ---------------
__global__ void proj_kernel(const float* __restrict__ y, const float* __restrict__ w,
                            const float* __restrict__ bias, const float* __restrict__ x,
                            float* __restrict__ out) {
  int idx = blockIdx.x * blockDim.x + threadIdx.x;       // 32*32*511
  if (idx >= NB * NC * LL) return;
  int t = idx % LL;
  int pb = idx / LL;
  int p = pb & 31;
  int b = pb >> 5;
  float acc = bias[p];
  const float* yr = y + (size_t)b * NC * LL + t;
  const float* wr = w + p * NC;
  #pragma unroll
  for (int o = 0; o < NC; o++) acc += wr[o] * yr[(size_t)o * LL];
  out[idx] = x[idx] + acc;
}

extern "C" void kernel_launch(void* const* d_in, const int* in_sizes, int n_in,
                              void* d_out, int out_size, void* d_ws, size_t ws_size,
                              hipStream_t stream) {
  const float* x      = (const float*)d_in[0];
  const float* norm_w = (const float*)d_in[1];
  const float* norm_b = (const float*)d_in[2];
  const float* qkv_w  = (const float*)d_in[3];
  const float* qkv_b  = (const float*)d_in[4];
  const float* proj_w = (const float*)d_in[5];
  const float* proj_b = (const float*)d_in[6];
  float* out = (float*)d_out;

  // workspace partition (all offsets multiples of 256B)
  char* p = (char*)d_ws;
  float*     h     = (float*)p;      p += (size_t)NB * NC * LL * 4;   //  2,093,056
  _Float16*  qh    = (_Float16*)p;   p += (size_t)NB * NC * LP * 2;   //  1,048,576
  _Float16*  kh    = (_Float16*)p;   p += (size_t)NB * NC * LP * 2;
  _Float16*  vh    = (_Float16*)p;   p += (size_t)NB * NC * LP * 2;
  float*     S     = (float*)p;      p += (size_t)NB * LP * LP * 4;   // 33,554,432
  _Float16*  attnh = (_Float16*)p;   p += (size_t)NB * LP * LP * 2;   // 16,777,216
  _Float16*  Ash   = (_Float16*)p;   p += (size_t)NB * NC * RW * 2;   //  2,097,152
  uint32_t*  Aswz  = (uint32_t*)p;   p += (size_t)2 * 512 * 32 * 8 * 4; // 1,048,576
  float*     y     = (float*)p;      p += (size_t)NB * NC * LL * 4;

  // 0) zero the padded shift image (pads must be exactly 0)
  zero_f16_kernel<<<512, 256, 0, stream>>>((float4*)Ash, (NB * NC * RW * 2) / 16);
  // 1) group norm
  gn_kernel<<<NB * NC, 256, 0, stream>>>(x, norm_w, norm_b, h);
  // 2) qkv (q pre-scaled by d^-0.5), padded f16
  qkv_kernel<<<(NB * 96 * LP) / 256, 256, 0, stream>>>(h, qkv_w, qkv_b, qh, kh, vh);
  // 3) attention scores (WMMA): 32 batches x 32x32 tiles, 4 waves/block
  s_kernel<<<8192, 128, 0, stream>>>(qh, kh, S);
  // 4) softmax -> f16 attn
  softmax_kernel<<<NB * LP, 128, 0, stream>>>(S, attnh);
  // 5) attn*v (WMMA) -> padded shift image
  av_kernel<<<512, 128, 0, stream>>>(attnh, vh, Ash);
  // 6) pre-swizzle unshifted operand into WMMA-A fragment layout
  aswz_kernel<<<128, 256, 0, stream>>>(Ash, Aswz);
  // 7) self-convolution: 511 shift-GEMMs (dominant 17.1 GFLOP, WMMA + async LDS)
  selfconv_kernel<<<LL, 128, 0, stream>>>(Ash, Aswz, y);
  // 8) projection + residual
  proj_kernel<<<(NB * NC * LL + 255) / 256, 256, 0, stream>>>(y, proj_w, proj_b, x, out);
  (void)in_sizes; (void)n_in; (void)out_size; (void)ws_size;
}
